// OuterProductMean_2199023255609
// MI455X (gfx1250) — matrix operand order repaired and checked by reading, not compiled, not run on hardware
//
#include <hip/hip_runtime.h>
#include <hip/hip_bf16.h>

// ---------------------------------------------------------------------------
// Problem constants:  B=1, NR=384, M=4, NL=128, IN_C=384, MID_C=32, OUT_C=128
// Output: [512, 512, 128] f32
// ---------------------------------------------------------------------------

typedef __attribute__((ext_vector_type(16))) __bf16 v16bf;
typedef __attribute__((ext_vector_type(8)))  float  v8f;

union ABf { uint4 u[2]; v16bf v; };

// A operand: 16x32 bf16 row-major tile; ISA 16-bit A layout:
//   lane<16 : row r, K 0..7 (chunk0) and K 16..23 (chunk1 = +32B)
//   lane>=16: row r, K 8..15 and K 24..31
static __device__ __forceinline__ v16bf loadA(const __hip_bfloat16* tile, int r, int hf) {
  const char* at = (const char*)tile + (r << 6) + (hf << 4);
  ABf a;
  a.u[0] = *(const uint4*)(at);
  a.u[1] = *(const uint4*)(at + 32);
  return a.v;
}

// B operand: factor stored row-major [n][32]; ISA 16-bit B layout:
//   lane<16 : col n, K 0..7 and K 16..23;  lane>=16: col n, K 8..15 and K 24..31
static __device__ __forceinline__ v16bf loadB(const __hip_bfloat16* tile, int r, int hf) {
  const char* bt = (const char*)tile + (r << 6) + (hf << 4);
  ABf b;
  b.u[0] = *(const uint4*)(bt);
  b.u[1] = *(const uint4*)(bt + 32);
  return b.v;
}

// Coalesced LDS -> global writeback of one 16x16x32 slab (2048 float4 total).
static __device__ __forceinline__ void writeback(
    const float* lds, float* __restrict__ out,
    int base, int si, int sj, int o0p, int tid)
{
  #pragma unroll
  for (int rr = 0; rr < 8; ++rr) {
    int f = tid + rr * 256;
    int pair = f >> 3, q = f & 7;
    float4 val = *(const float4*)&lds[pair * 36 + (q << 2)];
    *(float4*)&out[base + (pair >> 4) * si + (pair & 15) * sj + o0p + (q << 2)] = val;
  }
}

// ---------------------------------------------------------------------------
// Phase 1a: 1D projections.  1792 rows x 32 channels, K=384 dot products.
// ---------------------------------------------------------------------------
__global__ __launch_bounds__(256) void proj_kernel(
    const float* __restrict__ rec, const float* __restrict__ lig,
    const float* __restrict__ rl1_w, const float* __restrict__ rl1_b,
    const float* __restrict__ rl2_w, const float* __restrict__ rl2_b,
    const float* __restrict__ ll1_w, const float* __restrict__ ll1_b,
    const float* __restrict__ ll2_w, const float* __restrict__ ll2_b,
    float* __restrict__ rI, float* __restrict__ rJ,
    float* __restrict__ lI, float* __restrict__ lJ,
    __hip_bfloat16* __restrict__ rj_bf, __hip_bfloat16* __restrict__ lj_bf)
{
  int g = blockIdx.x * 256 + threadIdx.x;
  if (g >= 1792 * 32) return;
  int c = g & 31, row = g >> 5;

  const float* x; const float* w; float bb; int kind, rloc;
  if (row < 384)       { rloc = row;        x = rec + rloc * 384; w = rl1_w + c * 384; bb = rl1_b[c]; kind = 0; }
  else if (row < 768)  { rloc = row - 384;  x = rec + rloc * 384; w = rl2_w + c * 384; bb = rl2_b[c]; kind = 1; }
  else if (row < 1280) { rloc = row - 768;  x = lig + rloc * 384; w = ll1_w + c * 384; bb = ll1_b[c]; kind = 2; }
  else                 { rloc = row - 1280; x = lig + rloc * 384; w = ll2_w + c * 384; bb = ll2_b[c]; kind = 3; }

  float acc = bb;
  #pragma unroll 8
  for (int k = 0; k < 384; ++k) acc = fmaf(x[k], w[k], acc);

  int idx = rloc * 32 + c;
  if (kind == 0)      rI[idx] = acc;
  else if (kind == 1) { rJ[idx] = acc; rj_bf[idx] = __float2bfloat16(acc); }
  else if (kind == 2) lI[idx] = acc;
  else                { lJ[idx] = acc; lj_bf[idx] = __float2bfloat16(acc); }
}

// ---------------------------------------------------------------------------
// Phase 1b: means over the m=4 ligand copies -> bf16 B-side factors.
// ---------------------------------------------------------------------------
__global__ __launch_bounds__(256) void mean_kernel(
    const float* __restrict__ lI, const float* __restrict__ lJ,
    __hip_bfloat16* __restrict__ lim_bf, __hip_bfloat16* __restrict__ ljm_bf)
{
  int t = blockIdx.x * 256 + threadIdx.x;
  if (t >= 128 * 32) return;
  int j = t >> 5, c = t & 31;
  float si = 0.f, sj = 0.f;
  #pragma unroll
  for (int m = 0; m < 4; ++m) {
    si += lI[(m * 128 + j) * 32 + c];
    sj += lJ[(m * 128 + j) * 32 + c];
  }
  lim_bf[t] = __float2bfloat16(si * 0.25f);
  ljm_bf[t] = __float2bfloat16(sj * 0.25f);
}

// ---------------------------------------------------------------------------
// Phase 1c: A[row,o,y] = sum_x src[row,x] * W[o, x*32 + y], stored bf16 in
// WMMA-friendly tiles: dst[(row/16)][o][row%16][y]  (16x32 row-major tiles).
// ---------------------------------------------------------------------------
__global__ __launch_bounds__(256) void abuild_kernel(
    const float* __restrict__ src, const float* __restrict__ W,
    __hip_bfloat16* __restrict__ dst, int nrows)
{
  int g = blockIdx.x * 256 + threadIdx.x;
  int y = g & 31;
  int o = (g >> 5) & 127;
  int row = g >> 12;
  if (row >= nrows) return;
  const float* s  = src + row * 32;
  const float* wp = W + o * 1024 + y;
  float acc = 0.f;
  #pragma unroll
  for (int x = 0; x < 32; ++x) acc = fmaf(s[x], wp[x * 32], acc);
  dst[(((row >> 4) * 128 + o) << 9) + ((row & 15) << 5) + y] = __float2bfloat16(acc);
}

// ---------------------------------------------------------------------------
// Phase 2: one workgroup = one 16x16 (i,j) tile, all 128 output channels.
// 8 waves x 4 channels per pass, 4 passes.  Quadrant split is hoisted fully
// outside the hot loops: each loop body is branch-free straight-line code
// (2x global_load_b128 -> v_wmma_f32_16x16x32_bf16 -> ds stores).
// ---------------------------------------------------------------------------
__global__ __launch_bounds__(256) void opm_wmma_kernel(
    const __hip_bfloat16* __restrict__ Arr, const __hip_bfloat16* __restrict__ Arl,
    const __hip_bfloat16* __restrict__ Alr, const __hip_bfloat16* __restrict__ All,
    const __hip_bfloat16* __restrict__ rj_bf, const __hip_bfloat16* __restrict__ ljm_bf,
    const __hip_bfloat16* __restrict__ lim_bf, const __hip_bfloat16* __restrict__ lj_bf,
    const float* __restrict__ rr_b, const float* __restrict__ rl_b,
    const float* __restrict__ lr_b, const float* __restrict__ ll_b,
    float* __restrict__ out)
{
  __shared__ float lds[256 * 36];          // 16x16 pairs x 32 o (pitch 36) = 36 KB

  const int tid  = threadIdx.x;
  const int lane = tid & 31;
  const int wave = tid >> 5;
  const int r    = lane & 15;
  const int hf   = lane >> 4;

  int b = blockIdx.x;
  int quad, it, jt;
  if (b < 576)      { quad = 0; it = b / 24;            jt = b % 24; }
  else if (b < 768) { quad = 1; b -= 576; it = b >> 3;  jt = b & 7;  }
  else if (b < 960) { quad = 2; b -= 768; it = b >> 3;  jt = b & 7;  }
  else              { quad = 3; b -= 960; it = b >> 3;  jt = b & 7;  }

  const __hip_bfloat16* Abase; const __hip_bfloat16* Bt0; const float* bias;
  int base, si, sj;
  if (quad == 0) {        // rr: rows i (res), cols j (res)
    Abase = Arr + it * 65536; Bt0 = rj_bf + jt * 512; bias = rr_b;
    base = (it * 16 * 512 + jt * 16) * 128;            si = 65536; sj = 128;
  } else if (quad == 1) { // rl: rows i (res), cols 384+j (lig mean)
    Abase = Arl + it * 65536; Bt0 = ljm_bf + jt * 512; bias = rl_b;
    base = (it * 16 * 512 + 384 + jt * 16) * 128;      si = 65536; sj = 128;
  } else if (quad == 2) { // lr: output transposed -> rows 384+j, cols i
    Abase = Alr + it * 65536; Bt0 = lim_bf + jt * 512; bias = lr_b;
    base = ((384 + jt * 16) * 512 + it * 16) * 128;    si = 128;   sj = 65536;
  } else {                // ll: rows 384+i, cols 384+j, mean over m=4
    Abase = All + it * 65536; Bt0 = lj_bf + jt * 512;  bias = ll_b;
    base = ((384 + it * 16) * 512 + 384 + jt * 16) * 128; si = 65536; sj = 128;
  }

  const int col = r;                       // D layout: N = lane%16
  const int rw0 = hf * 8;                  // D layout: M = v + 8*(lane/16)

  if (quad != 3) {
    // ------------------------- single-factor path -------------------------
    v16bf b0 = loadB(Bt0, r, hf);
    for (int pass = 0; pass < 4; ++pass) {
      float4 bo4 = *(const float4*)(bias + pass * 32 + wave * 4);
      #pragma unroll
      for (int k = 0; k < 4; ++k) {
        int ol = wave * 4 + k;
        int o  = pass * 32 + ol;
        v16bf a = loadA(Abase + o * 512, r, hf);
        v8f c = {};
        c = __builtin_amdgcn_wmma_f32_16x16x32_bf16(
                false, a, false, b0, (short)0, c, false, false);
        float bo = (k == 0) ? bo4.x : (k == 1) ? bo4.y : (k == 2) ? bo4.z : bo4.w;
        #pragma unroll
        for (int v = 0; v < 8; ++v)
          lds[((rw0 + v) * 16 + col) * 36 + ol] = c[v] + bo;
      }
      __syncthreads();
      writeback(lds, out, base, si, sj, pass * 32, tid);
      __syncthreads();
    }
  } else {
    // ---------------- ll path: 4 chained WMMAs over m, x0.25 --------------
    v16bf b0 = loadB(Bt0 + 0 * 4096, r, hf);   // lj_bf m-stride = 128*32
    v16bf b1 = loadB(Bt0 + 1 * 4096, r, hf);
    v16bf b2 = loadB(Bt0 + 2 * 4096, r, hf);
    v16bf b3 = loadB(Bt0 + 3 * 4096, r, hf);
    for (int pass = 0; pass < 4; ++pass) {
      float4 bo4 = *(const float4*)(bias + pass * 32 + wave * 4);
      #pragma unroll
      for (int k = 0; k < 4; ++k) {
        int ol = wave * 4 + k;
        int o  = pass * 32 + ol;
        const __hip_bfloat16* ap = Abase + o * 512;   // A_ll m-stride = 524288
        v8f c = {};
        v16bf a;
        a = loadA(ap + 0 * 524288, r, hf);
        c = __builtin_amdgcn_wmma_f32_16x16x32_bf16(false, a, false, b0, (short)0, c, false, false);
        a = loadA(ap + 1 * 524288, r, hf);
        c = __builtin_amdgcn_wmma_f32_16x16x32_bf16(false, a, false, b1, (short)0, c, false, false);
        a = loadA(ap + 2 * 524288, r, hf);
        c = __builtin_amdgcn_wmma_f32_16x16x32_bf16(false, a, false, b2, (short)0, c, false, false);
        a = loadA(ap + 3 * 524288, r, hf);
        c = __builtin_amdgcn_wmma_f32_16x16x32_bf16(false, a, false, b3, (short)0, c, false, false);
        float bo = (k == 0) ? bo4.x : (k == 1) ? bo4.y : (k == 2) ? bo4.z : bo4.w;
        #pragma unroll
        for (int v = 0; v < 8; ++v)
          lds[((rw0 + v) * 16 + col) * 36 + ol] = c[v] * 0.25f + bo;
      }
      __syncthreads();
      writeback(lds, out, base, si, sj, pass * 32, tid);
      __syncthreads();
    }
  }
}

// ---------------------------------------------------------------------------
// Host launcher
// ---------------------------------------------------------------------------
extern "C" void kernel_launch(void* const* d_in, const int* in_sizes, int n_in,
                              void* d_out, int out_size, void* d_ws, size_t ws_size,
                              hipStream_t stream)
{
  (void)in_sizes; (void)n_in; (void)out_size; (void)ws_size;

  const float* rec   = (const float*)d_in[0];   // [1,384,384]
  const float* lig   = (const float*)d_in[1];   // [1,4,128,384]
  // d_in[2] = pw_rep (unused; reference returns only the update)
  const float* rl1_w = (const float*)d_in[3];
  const float* rl1_b = (const float*)d_in[4];
  const float* rl2_w = (const float*)d_in[5];
  const float* rl2_b = (const float*)d_in[6];
  const float* ll1_w = (const float*)d_in[7];
  const float* ll1_b = (const float*)d_in[8];
  const float* ll2_w = (const float*)d_in[9];
  const float* ll2_b = (const float*)d_in[10];
  const float* rr_w  = (const float*)d_in[11];
  const float* rr_b  = (const float*)d_in[12];
  const float* rl_w  = (const float*)d_in[13];
  const float* rl_b  = (const float*)d_in[14];
  const float* lr_w  = (const float*)d_in[15];
  const float* lr_b  = (const float*)d_in[16];
  const float* ll_w  = (const float*)d_in[17];
  const float* ll_b  = (const float*)d_in[18];
  float* out = (float*)d_out;

  char* ws = (char*)d_ws;
  float* rI = (float*)(ws + 0);                              // 384x32 f32
  float* rJ = (float*)(ws + 49152);                          // 384x32 f32
  float* lI = (float*)(ws + 98304);                          // 512x32 f32
  float* lJ = (float*)(ws + 163840);                         // 512x32 f32
  __hip_bfloat16* rj_bf  = (__hip_bfloat16*)(ws + 229376);   // 384x32
  __hip_bfloat16* lj_bf  = (__hip_bfloat16*)(ws + 253952);   // 512x32
  __hip_bfloat16* lim_bf = (__hip_bfloat16*)(ws + 286720);   // 128x32
  __hip_bfloat16* ljm_bf = (__hip_bfloat16*)(ws + 294912);   // 128x32
  __hip_bfloat16* Arr = (__hip_bfloat16*)(ws + 303104);              // 24x128x512
  __hip_bfloat16* Arl = (__hip_bfloat16*)(ws + 303104 + 3145728);    // 24x128x512
  __hip_bfloat16* Alr = (__hip_bfloat16*)(ws + 303104 + 6291456);    // 24x128x512
  __hip_bfloat16* All = (__hip_bfloat16*)(ws + 303104 + 9437184);    // 32x128x512

  // Phase 1a: projections (1792 rows x 32 ch)
  proj_kernel<<<224, 256, 0, stream>>>(rec, lig,
      rl1_w, rl1_b, rl2_w, rl2_b, ll1_w, ll1_b, ll2_w, ll2_b,
      rI, rJ, lI, lJ, rj_bf, lj_bf);

  // Phase 1b: ligand means
  mean_kernel<<<16, 256, 0, stream>>>(lI, lJ, lim_bf, ljm_bf);

  // Phase 1c: per-channel A tensors (bf16, WMMA tile layout)
  abuild_kernel<<<6144, 256, 0, stream>>>(rI, rr_w, Arr, 384);
  abuild_kernel<<<6144, 256, 0, stream>>>(rI, rl_w, Arl, 384);
  abuild_kernel<<<6144, 256, 0, stream>>>(rJ, lr_w, Alr, 384);
  abuild_kernel<<<8192, 256, 0, stream>>>(lI, ll_w, All, 512);

  // Phase 2: 576 rr + 192 rl + 192 lr + 64 ll tiles = 1024 workgroups
  opm_wmma_kernel<<<1024, 256, 0, stream>>>(
      Arr, Arl, Alr, All, rj_bf, ljm_bf, lim_bf, lj_bf,
      rr_b, rl_b, lr_b, ll_b, out);
}
